// SingleHeadAttention_45114336477256
// MI455X (gfx1250) — matrix operand order repaired and checked by reading, not compiled, not run on hardware
//
#include <hip/hip_runtime.h>
#include <hip/hip_bf16.h>

// ---------------------------------------------------------------------------
// Single-head causal attention, B=8 S=2048 D=1024 H=64, fp32 in/out.
//   pack_x   : x fp32 -> bf16 (packed pairs), read by projection
//   pack_w   : Wq/Wk/Wv fp32 -> bf16 in exact WMMA B-fragment memory order
//   qkv_proj : bf16 WMMA projection, M=32 per wave; V stored transposed
//   attn     : flash attention, 2-wave split-K per query tile, bf16 WMMA
//              QK^T and PV, exp2-domain online softmax, LDS merge
// ---------------------------------------------------------------------------

#define BATCH 8
#define SEQ   2048
#define DIN   1024
#define HEAD  64
#define TOT   (BATCH * SEQ)          // 16384 rows
#define WP_DW 32768                  // packed dwords per weight matrix (D*H/2)

typedef __attribute__((ext_vector_type(16))) __bf16 v16bf;
typedef __attribute__((ext_vector_type(8)))  float  v8f;

union FragB {
    unsigned u[8];
    v16bf    v;
};

__device__ inline unsigned short f32_to_bf16_1(float f) {
    unsigned u = __float_as_uint(f);
    unsigned r = u + 0x7FFFu + ((u >> 16) & 1u);   // RNE
    return (unsigned short)(r >> 16);
}
__device__ inline unsigned pack_bf16(float lo, float hi) {
#if __has_builtin(__builtin_amdgcn_cvt_pk_bf16_f32)
    auto r = __builtin_amdgcn_cvt_pk_bf16_f32(lo, hi);   // v_cvt_pk_bf16_f32
    unsigned out;
    __builtin_memcpy(&out, &r, sizeof(out));
    return out;
#else
    return (unsigned)f32_to_bf16_1(lo) | ((unsigned)f32_to_bf16_1(hi) << 16);
#endif
}

// 16-bit A-matrix (16x32) layout: lane m=L%16, half=L/16; VGPR v holds K pair at:
__device__ inline int a_k0(int v, int half) {
    return (v < 4) ? (2 * v + 8 * half) : (16 + 2 * (v - 4) + 8 * half);
}
// 16-bit B-matrix (32x16) layout: lane n=L%16; same K order as the A layout
__device__ inline int b_k0(int v, int half) {
    return a_k0(v, half);
}

// ---------------------------------------------------------------------------
// x fp32 -> packed bf16 pairs (element order preserved)
// ---------------------------------------------------------------------------
__global__ __launch_bounds__(256) void pack_x_kernel(const float* __restrict__ x,
                                                     unsigned* __restrict__ xb) {
    size_t i = (size_t)blockIdx.x * 256 + threadIdx.x;   // one per 4 floats
    float4 f = ((const float4*)x)[i];
    uint2 o;
    o.x = pack_bf16(f.x, f.y);
    o.y = pack_bf16(f.z, f.w);
    ((uint2*)xb)[i] = o;
}

// ---------------------------------------------------------------------------
// Pack Wq/Wk/Wv into B-fragment memory order:
//   dword index (per matrix) = ((ks*4 + j)*32 + lane)*8 + v
//   contents = { W[ks*32 + b_k0(v,half)][j*16+lanelo], W[..+1][..] } as bf16 pair
// ---------------------------------------------------------------------------
__global__ __launch_bounds__(256) void pack_w_kernel(const float* __restrict__ Wq,
                                                     const float* __restrict__ Wk,
                                                     const float* __restrict__ Wv,
                                                     unsigned* __restrict__ Wp) {
    int idx = blockIdx.x * 256 + threadIdx.x;            // 0 .. 3*WP_DW-1
    int mat = idx >> 15;
    int rem = idx & (WP_DW - 1);
    const float* W = (mat == 0) ? Wq : (mat == 1) ? Wk : Wv;
    int v    = rem & 7;
    int lane = (rem >> 3) & 31;
    int j    = (rem >> 8) & 3;
    int ks   = rem >> 10;
    int half = lane >> 4, lanelo = lane & 15;
    int kr   = ks * 32 + b_k0(v, half);
    int col  = j * 16 + lanelo;
    Wp[idx] = pack_bf16(W[kr * HEAD + col], W[(kr + 1) * HEAD + col]);
}

// ---------------------------------------------------------------------------
// Projection: one wave computes 32 rows x 64 cols of one matrix (blockIdx.y).
// mat 0 -> Qb [TOT][64], mat 1 -> Kb [TOT][64], mat 2 -> Vt [64][TOT]
// ---------------------------------------------------------------------------
__global__ __launch_bounds__(256) void qkv_proj_kernel(
    const unsigned* __restrict__ xb,     // bf16-packed x, [TOT][DIN/2] dwords
    const unsigned* __restrict__ Wp,     // 3 packed weight matrices
    const float* __restrict__ bq, const float* __restrict__ bk, const float* __restrict__ bv,
    unsigned short* __restrict__ Qb,
    unsigned short* __restrict__ Kb,
    unsigned short* __restrict__ Vt)
{
    const int lane   = threadIdx.x & 31;
    const int wave   = threadIdx.x >> 5;
    const int half   = lane >> 4;
    const int lanelo = lane & 15;
    const int mat    = blockIdx.y;
    const int tile   = blockIdx.x * 8 + wave;            // 512 tiles of 32 rows
    const long row0  = (long)tile * 32;
    const unsigned* W = Wp + (size_t)mat * WP_DW;

    v8f acc[4][2];
#pragma unroll
    for (int j = 0; j < 4; ++j) { acc[j][0] = {}; acc[j][1] = {}; }

    for (int ks = 0; ks < DIN / 32; ++ks) {
        FragB a0, a1;
        const unsigned* x0 = xb + (row0 + lanelo) * (DIN / 2) + ks * 16;
        const unsigned* x1 = x0 + 16 * (DIN / 2);
#pragma unroll
        for (int v = 0; v < 8; ++v) {
            int k0h = a_k0(v, half) >> 1;                // dword index, k0 is even
            a0.u[v] = x0[k0h];
            a1.u[v] = x1[k0h];
        }
#pragma unroll
        for (int j = 0; j < 4; ++j) {
            FragB bm;
            const unsigned* bp = W + ((size_t)(ks * 4 + j) * 32 + lane) * 8;
#pragma unroll
            for (int v = 0; v < 8; ++v) bm.u[v] = bp[v]; // 8 contiguous dwords -> b128
            acc[j][0] = __builtin_amdgcn_wmma_f32_16x16x32_bf16(false, a0.v, false, bm.v, (short)0, acc[j][0], false, false);
            acc[j][1] = __builtin_amdgcn_wmma_f32_16x16x32_bf16(false, a1.v, false, bm.v, (short)0, acc[j][1], false, false);
        }
    }

    const float* bias = (mat == 0) ? bq : (mat == 1) ? bk : bv;
#pragma unroll
    for (int j = 0; j < 4; ++j) {
        float bb = bias[j * 16 + lanelo];
#pragma unroll
        for (int mt = 0; mt < 2; ++mt)
#pragma unroll
            for (int i = 0; i < 8; ++i) {
                long r = row0 + mt * 16 + i + 8 * half;
                unsigned short val = f32_to_bf16_1(acc[j][mt][i] + bb);
                if (mat == 0)      Qb[r * HEAD + j * 16 + lanelo] = val;
                else if (mat == 1) Kb[r * HEAD + j * 16 + lanelo] = val;
                else               Vt[(size_t)(j * 16 + lanelo) * TOT + r] = val;
            }
    }
}

// ---------------------------------------------------------------------------
// Flash attention: one 2-wave workgroup per (batch, 16-query tile).
// Wave w handles key tiles kt == w (mod 2); partials merged through LDS.
// ---------------------------------------------------------------------------
__global__ __launch_bounds__(64) void attn_kernel(
    const unsigned short* __restrict__ Qb,
    const unsigned short* __restrict__ Kb,
    const unsigned short* __restrict__ Vt,   // [64][TOT]
    float* __restrict__ out)
{
    __shared__ unsigned short ldsP[2][16 * 32];   // per-wave P staging
    __shared__ float ldsO[16 * 64];               // wave1 partial O
    __shared__ float ldsM[16], ldsL[16];          // wave1 partial m, l

    const int lane   = threadIdx.x & 31;
    const int wv     = threadIdx.x >> 5;          // 0 or 1
    const int half   = lane >> 4;
    const int lanelo = lane & 15;
    const int qt     = blockIdx.x & (SEQ / 16 - 1);
    const int b      = blockIdx.x >> 7;
    const long qrow0 = (long)b * SEQ + qt * 16;
    const float SC   = 0.125f * 1.44269504f;      // 1/sqrt(64) * log2(e)

    // Resident Q fragments (A-layout), h = 0..63 over two K-steps
    FragB qa[2];
#pragma unroll
    for (int ks = 0; ks < 2; ++ks) {
        const unsigned short* qrow = Qb + (qrow0 + lanelo) * HEAD + ks * 32;
#pragma unroll
        for (int v = 0; v < 8; ++v)
            qa[ks].u[v] = *(const unsigned*)(qrow + a_k0(v, half));
    }

    v8f acc[4];
#pragma unroll
    for (int j = 0; j < 4; ++j) acc[j] = {};
    float m_row[8], l_row[8];
#pragma unroll
    for (int i = 0; i < 8; ++i) { m_row[i] = -1e30f; l_row[i] = 0.0f; }

    const int qmax = qt * 16 + 15;
    for (int kt = wv; kt * 32 <= qmax; kt += 2) {        // wave-uniform trip count
        const long key0 = (long)b * SEQ + kt * 32;

        // ----- scores: two 16x16 tiles, K-dim = h = 64
        v8f s[2];
#pragma unroll
        for (int t = 0; t < 2; ++t) {
            const unsigned short* krow = Kb + (key0 + t * 16 + lanelo) * HEAD;
            v8f st = {};
#pragma unroll
            for (int ks = 0; ks < 2; ++ks) {
                FragB kb;
#pragma unroll
                for (int v = 0; v < 8; ++v)
                    kb.u[v] = *(const unsigned*)(krow + b_k0(v, half) + ks * 32);
                st = __builtin_amdgcn_wmma_f32_16x16x32_bf16(false, qa[ks].v, false, kb.v, (short)0, st, false, false);
            }
            s[t] = st;
        }

        // ----- scale into exp2 domain + causal mask (lane selects, EXEC untouched)
#pragma unroll
        for (int i = 0; i < 8; ++i) {
            int qg  = qt * 16 + i + 8 * half;
            int kg0 = kt * 32 + lanelo;
            s[0][i] = (kg0      <= qg) ? s[0][i] * SC : -1e30f;
            s[1][i] = (kg0 + 16 <= qg) ? s[1][i] * SC : -1e30f;
        }

        // ----- online softmax: each row's 32 scores live in one 16-lane half
        float p0[8], p1[8], alpha[8];
#pragma unroll
        for (int i = 0; i < 8; ++i) {
            float cand = fmaxf(s[0][i], s[1][i]);
#pragma unroll
            for (int d = 1; d < 16; d <<= 1) cand = fmaxf(cand, __shfl_xor(cand, d, 32));
            float nm = fmaxf(m_row[i], cand);
            alpha[i] = exp2f(m_row[i] - nm);
            m_row[i] = nm;
            p0[i] = exp2f(s[0][i] - nm);
            p1[i] = exp2f(s[1][i] - nm);
            float rs = p0[i] + p1[i];
#pragma unroll
            for (int d = 1; d < 16; d <<= 1) rs += __shfl_xor(rs, d, 32);
            l_row[i] = l_row[i] * alpha[i] + rs;
        }
#pragma unroll
        for (int j = 0; j < 4; ++j)
#pragma unroll
            for (int i = 0; i < 8; ++i) acc[j][i] *= alpha[i];

        // ----- re-layout P (C-layout) -> A-layout via per-wave LDS tile
#pragma unroll
        for (int i = 0; i < 8; ++i) {
            int m = i + 8 * half;
            ldsP[wv][m * 32 + lanelo]      = f32_to_bf16_1(p0[i]);
            ldsP[wv][m * 32 + 16 + lanelo] = f32_to_bf16_1(p1[i]);
        }
        asm volatile("s_wait_dscnt 0" ::: "memory");
        FragB pa;
#pragma unroll
        for (int v = 0; v < 8; ++v)
            pa.u[v] = *(const unsigned*)(&ldsP[wv][lanelo * 32 + a_k0(v, half)]);

        // ----- O += P(16x32) * V(32x64); V^T gives aligned dword pairs
#pragma unroll
        for (int j = 0; j < 4; ++j) {
            FragB vb;
            const unsigned short* vcol = Vt + (size_t)(j * 16 + lanelo) * TOT + key0;
#pragma unroll
            for (int v = 0; v < 8; ++v)
                vb.u[v] = *(const unsigned*)(vcol + b_k0(v, half));
            acc[j] = __builtin_amdgcn_wmma_f32_16x16x32_bf16(false, pa.v, false, vb.v, (short)0, acc[j], false, false);
        }
    }

    // ----- split-K merge: wave1 publishes partials, wave0 combines + stores
    if (wv == 1) {
#pragma unroll
        for (int i = 0; i < 8; ++i) {
            int m = i + 8 * half;
            if (lanelo == 0) { ldsM[m] = m_row[i]; ldsL[m] = l_row[i]; }
#pragma unroll
            for (int j = 0; j < 4; ++j)
                ldsO[m * 64 + j * 16 + lanelo] = acc[j][i];
        }
    }
    __syncthreads();
    if (wv == 0) {
#pragma unroll
        for (int i = 0; i < 8; ++i) {
            int   m  = i + 8 * half;
            float m1 = ldsM[m];
            float l1 = ldsL[m];
            float mm = fmaxf(m_row[i], m1);
            float a0 = exp2f(m_row[i] - mm);
            float a1 = exp2f(m1 - mm);
            float inv = 1.0f / (l_row[i] * a0 + l1 * a1);
#pragma unroll
            for (int j = 0; j < 4; ++j) {
                long off = (qrow0 + m) * HEAD + j * 16 + lanelo;
                out[off] = (acc[j][i] * a0 + ldsO[m * 64 + j * 16 + lanelo] * a1) * inv;
            }
        }
    }
}

// ---------------------------------------------------------------------------
extern "C" void kernel_launch(void* const* d_in, const int* in_sizes, int n_in,
                              void* d_out, int out_size, void* d_ws, size_t ws_size,
                              hipStream_t stream) {
    const float* x  = (const float*)d_in[0];
    const float* Wq = (const float*)d_in[1];
    const float* bq = (const float*)d_in[2];
    const float* Wk = (const float*)d_in[3];
    const float* bk = (const float*)d_in[4];
    const float* Wv = (const float*)d_in[5];
    const float* bv = (const float*)d_in[6];
    float* out = (float*)d_out;

    char* ws = (char*)d_ws;
    unsigned*       xb = (unsigned*)ws;                                   // 32 MB
    unsigned short* Qb = (unsigned short*)(ws + (size_t)TOT * DIN * 2);   // 2 MB
    unsigned short* Kb = Qb + (size_t)TOT * HEAD;                         // 2 MB
    unsigned short* Vt = Kb + (size_t)TOT * HEAD;                         // 2 MB
    unsigned*       Wp = (unsigned*)(Vt + (size_t)TOT * HEAD);            // 384 KB

    pack_x_kernel<<<(size_t)TOT * DIN / 4 / 256, 256, 0, stream>>>(x, xb);
    pack_w_kernel<<<3 * WP_DW / 256, 256, 0, stream>>>(Wq, Wk, Wv, Wp);

    dim3 pgrid(TOT / 32 / 8, 3);
    qkv_proj_kernel<<<pgrid, 256, 0, stream>>>(xb, Wp, bq, bk, bv, Qb, Kb, Vt);

    // one 2-wave workgroup per (batch, 16-query tile), split-K over key tiles
    attn_kernel<<<BATCH * (SEQ / 16), 64, 0, stream>>>(Qb, Kb, Vt, out);
}
